// CrossHazardInteractionLayer_62156766708197
// MI455X (gfx1250) — matrix operand order repaired and checked by reading, not compiled, not run on hardware
//
#include <hip/hip_runtime.h>
#include <math.h>

// Problem constants (reference: H=7, B=8, S=2048, D=768, K=64, THR=0.05)
#define NH 7
#define BS 16384           // B*S
#define ND 768
#define NK 64
#define THRG 0.05f

#define WROWS 32           // rows per wave (1-wave workgroups, barrier-free)
#define HPAD 72            // LDS halves per h row (64 + 8 pad)
#define KC1 (ND / 32)      // 24 K-chunks for GEMM1
#define NT2 (ND / 16)      // 48 output column tiles for GEMM2

typedef __attribute__((ext_vector_type(16))) _Float16 v16h;
typedef __attribute__((ext_vector_type(8)))  _Float16 v8h;
typedef __attribute__((ext_vector_type(8)))  float    v8f;

union AB { v16h v; v8h h[2]; };

__device__ __forceinline__ v8f wmma_f16(v16h a, v16h b, v8f c) {
  return __builtin_amdgcn_wmma_f32_16x16x32_f16(false, a, false, b, (short)0, c, false, false);
}

// CDNA5 hardware tanh (transcendental VALU, co-executes with WMMA).
__device__ __forceinline__ float tanh_fast(float v) {
#if __has_builtin(__builtin_amdgcn_tanhf)
  return __builtin_amdgcn_tanhf(v);
#else
  float r;
  asm volatile("v_tanh_f32 %0, %1\n\tv_nop\n\tv_nop" : "=v"(r) : "v"(v));
  return r;
#endif
}

__device__ __forceinline__ float gelu_tanh(float z) {
  const float inner = z * (0.79788456080286535588f + 0.03567740814183397f * z * z);
  return 0.5f * z * (1.0f + tanh_fast(inner));
}

// ---------------------------------------------------------------------------
// Weight packing into WMMA-B lane order (tile = 32K x 16N; lane holds
// N = lane&15, K = (e>>3)*16 + (lane>>4)*8 + (e&7); 16 halves contiguous per lane).
// ---------------------------------------------------------------------------
__global__ void pack_w1(const float* __restrict__ W1, _Float16* __restrict__ dst) {
  const size_t total = (size_t)NH * NH * ND * NK;
  size_t o = (size_t)blockIdx.x * blockDim.x + threadIdx.x;
  if (o >= total) return;
  const int e    = (int)(o & 15);
  const int lane = (int)((o >> 4) & 31);
  const int nt   = (int)((o >> 9) & 3);
  const int kc   = (int)((o >> 11) % KC1);
  const int pair = (int)((o >> 11) / KC1);
  const int d = kc * 32 + (e >> 3) * 16 + (lane >> 4) * 8 + (e & 7);
  const int k = nt * 16 + (lane & 15);
  dst[o] = (_Float16)W1[((size_t)pair * ND + d) * NK + k];
}

__global__ void pack_w2(const float* __restrict__ W2, _Float16* __restrict__ dst) {
  const size_t total = (size_t)NH * NH * ND * NK;
  size_t o = (size_t)blockIdx.x * blockDim.x + threadIdx.x;
  if (o >= total) return;
  const int e    = (int)(o & 15);
  const int lane = (int)((o >> 4) & 31);
  int q          = (int)(o >> 9);
  const int nt   = q % NT2;  q /= NT2;
  const int kc   = q & 1;
  const int pair = q >> 1;
  const int kk = kc * 32 + (e >> 3) * 16 + (lane >> 4) * 8 + (e & 7);
  const int n  = nt * 16 + (lane & 15);
  dst[o] = (_Float16)W2[((size_t)pair * NK + kk) * ND + n];
}

// gb2[t][d] = sum_s gate(s,t) * b2[s,t,d]
__global__ void gate_bias2(const float* __restrict__ M, const float* __restrict__ b2,
                           float* __restrict__ gb2) {
  const int t = blockIdx.x;
  for (int c = threadIdx.x; c < ND; c += blockDim.x) {
    float sum = 0.0f;
    #pragma unroll
    for (int s = 0; s < NH; ++s) {
      const float m = M[s * NH + t];
      const float g = (s != t && fabsf(m) > THRG) ? m : 0.0f;
      sum += g * b2[(size_t)(s * NH + t) * ND + c];
    }
    gb2[t * ND + c] = sum;
  }
}

// x fp32 -> f16 (176 MB result fits the 192 MB L2 for the 6x per-source reuse)
__global__ void cvt_x(const float* __restrict__ x, _Float16* __restrict__ xh) {
  const size_t i = ((size_t)blockIdx.x * blockDim.x + threadIdx.x) * 8;
  const float4 f0 = ((const float4*)(x + i))[0];
  const float4 f1 = ((const float4*)(x + i))[1];
  v8h h = { (_Float16)f0.x, (_Float16)f0.y, (_Float16)f0.z, (_Float16)f0.w,
            (_Float16)f1.x, (_Float16)f1.y, (_Float16)f1.z, (_Float16)f1.w };
  *(v8h*)(xh + i) = h;
}

// ---------------------------------------------------------------------------
// Fused kernel: 1 wave per workgroup, 32 rows, one target t, barrier-free.
// F16X path software-pipelines A/B register double-buffers across K-steps.
// Bias rides the GEMM1 accumulator init; residual + gb2 ride GEMM2's C.
// ---------------------------------------------------------------------------
template <bool F16X>
__global__ __launch_bounds__(32) void hazard_fused(
    const float* __restrict__ x, const float* __restrict__ M,
    const float* __restrict__ b1, const _Float16* __restrict__ pW1,
    const _Float16* __restrict__ pW2, const float* __restrict__ gb2,
    const _Float16* __restrict__ xh, float* __restrict__ out) {
  __shared__ __align__(16) _Float16 sh_h[6 * WROWS * HPAD];  // 27.6 KB

  const int t       = blockIdx.y;
  const int rowBase = blockIdx.x * WROWS;
  const int lane = threadIdx.x;
  const int lr   = lane & 15;
  const int hi   = lane >> 4;
  const int kb8  = hi * 8;

  float gv[NH];
  #pragma unroll
  for (int s = 0; s < NH; ++s) {
    const float m = M[s * NH + t];
    gv[s] = (s != t && fabsf(m) > THRG) ? m : 0.0f;
  }

  // ------------------- GEMM1 + bias + GELU + gate -> LDS h ------------------
  int slot = 0;
  #pragma unroll 1
  for (int s = 0; s < NH; ++s) {
    if (gv[s] == 0.0f) continue;                 // uniform branch
    const float g  = gv[s];
    const int pair = s * NH + t;

    // accumulators pre-loaded with the (column-only) bias broadcast
    const float* b1p = b1 + (size_t)pair * NK;
    v8f acc[2][4];
    #pragma unroll
    for (int nt = 0; nt < 4; ++nt) {
      const float bv = b1p[nt * 16 + lr];
      #pragma unroll
      for (int rb = 0; rb < 2; ++rb)
        #pragma unroll
        for (int v = 0; v < 8; ++v) acc[rb][nt][v] = bv;
    }

    const size_t rowA0   = (size_t)s * BS + rowBase + lr;
    const _Float16* xp   = F16X ? (xh + rowA0 * ND) : (const _Float16*)0;
    const float* xf      = x + rowA0 * ND;
    const _Float16* pWb  = pW1 + (size_t)pair * KC1 * 2048 + lane * 16;

    auto loadA16 = [&](int kc, AB (&A)[2]) {
      #pragma unroll
      for (int rb = 0; rb < 2; ++rb) {
        const _Float16* ap = xp + (size_t)rb * 16 * ND + kc * 32;
        A[rb].h[0] = *(const v8h*)(ap + kb8);
        A[rb].h[1] = *(const v8h*)(ap + 16 + kb8);
      }
    };
    auto loadA32 = [&](int kc, AB (&A)[2]) {
      #pragma unroll
      for (int rb = 0; rb < 2; ++rb) {
        const float* ap = xf + (size_t)rb * 16 * ND + kc * 32;
        const float4 f0 = *(const float4*)(ap + kb8);
        const float4 f1 = *(const float4*)(ap + kb8 + 4);
        const float4 f2 = *(const float4*)(ap + 16 + kb8);
        const float4 f3 = *(const float4*)(ap + 20 + kb8);
        v8h a0 = { (_Float16)f0.x, (_Float16)f0.y, (_Float16)f0.z, (_Float16)f0.w,
                   (_Float16)f1.x, (_Float16)f1.y, (_Float16)f1.z, (_Float16)f1.w };
        v8h a1 = { (_Float16)f2.x, (_Float16)f2.y, (_Float16)f2.z, (_Float16)f2.w,
                   (_Float16)f3.x, (_Float16)f3.y, (_Float16)f3.z, (_Float16)f3.w };
        A[rb].h[0] = a0;
        A[rb].h[1] = a1;
      }
    };
    auto loadB = [&](int kc, AB (&Bv)[4]) {
      const _Float16* bb = pWb + (size_t)kc * 2048;
      #pragma unroll
      for (int nt = 0; nt < 4; ++nt) {
        Bv[nt].h[0] = *(const v8h*)(bb + nt * 512);
        Bv[nt].h[1] = *(const v8h*)(bb + nt * 512 + 8);
      }
    };
    auto domm = [&](AB (&A)[2], AB (&Bv)[4]) {
      #pragma unroll
      for (int nt = 0; nt < 4; ++nt) {
        acc[0][nt] = wmma_f16(A[0].v, Bv[nt].v, acc[0][nt]);
        acc[1][nt] = wmma_f16(A[1].v, Bv[nt].v, acc[1][nt]);
      }
    };

    if (F16X) {
      // software-pipelined: double-buffered A and B registers
      AB Ab[2][2], Bb[2][4];
      loadA16(0, Ab[0]);  loadB(0, Bb[0]);
      #pragma unroll 1
      for (int kc = 0; kc < KC1 - 2; kc += 2) {
        loadA16(kc + 1, Ab[1]);  loadB(kc + 1, Bb[1]);
        domm(Ab[0], Bb[0]);
        loadA16(kc + 2, Ab[0]);  loadB(kc + 2, Bb[0]);
        domm(Ab[1], Bb[1]);
      }
      loadA16(KC1 - 1, Ab[1]);  loadB(KC1 - 1, Bb[1]);
      domm(Ab[0], Bb[0]);
      domm(Ab[1], Bb[1]);
    } else {
      #pragma unroll 1
      for (int kc = 0; kc < KC1; ++kc) {
        AB A[2], Bv[4];
        loadA32(kc, A);
        loadB(kc, Bv);
        domm(A, Bv);
      }
    }

    // epilogue: h = gate * gelu(acc) -> f16 LDS slot (wave-private)
    _Float16* hs = sh_h + slot * (WROWS * HPAD);
    #pragma unroll
    for (int nt = 0; nt < 4; ++nt) {
      #pragma unroll
      for (int rb = 0; rb < 2; ++rb) {
        #pragma unroll
        for (int v = 0; v < 8; ++v) {
          const float z = acc[rb][nt][v];
          hs[(rb * 16 + hi * 8 + v) * HPAD + nt * 16 + lr] = (_Float16)(g * gelu_tanh(z));
        }
      }
    }
    ++slot;
  }

  // ------------------- GEMM2: C preloaded with x[t] + gb2 -------------------
  const size_t tBase = ((size_t)t * BS + rowBase) * ND;
  const float* xt = x + tBase;
  float* ot       = out + tBase;
  const float* gb = gb2 + t * ND;

  #pragma unroll 1
  for (int nt = 0; nt < NT2; nt += 2) {           // 2 column tiles per pass
    v8f acc2[2][2];                               // [col j][row tile rb]
    #pragma unroll
    for (int j = 0; j < 2; ++j) {
      const int col   = (nt + j) * 16 + lr;
      const float gbv = gb[col];
      #pragma unroll
      for (int rb = 0; rb < 2; ++rb)
        #pragma unroll
        for (int v = 0; v < 8; ++v)
          acc2[j][rb][v] = xt[(size_t)(rb * 16 + hi * 8 + v) * ND + col] + gbv;
    }

    int sl = 0;
    #pragma unroll
    for (int s = 0; s < NH; ++s) {
      if (gv[s] == 0.0f) continue;
      const int pair = s * NH + t;
      const _Float16* hs  = sh_h + sl * (WROWS * HPAD);
      const _Float16* bp0 = pW2 + ((size_t)(pair * 2) * NT2 + nt) * 512 + lane * 16;
      #pragma unroll
      for (int kc = 0; kc < 2; ++kc) {
        AB Bq[2], Aq[2];
        const _Float16* bp = bp0 + (size_t)kc * NT2 * 512;
        #pragma unroll
        for (int j = 0; j < 2; ++j) {
          Bq[j].h[0] = *(const v8h*)(bp + j * 512);
          Bq[j].h[1] = *(const v8h*)(bp + j * 512 + 8);
        }
        #pragma unroll
        for (int rb = 0; rb < 2; ++rb) {
          const _Float16* ap = hs + (size_t)(rb * 16 + lr) * HPAD + kc * 32;
          Aq[rb].h[0] = *(const v8h*)(ap + kb8);
          Aq[rb].h[1] = *(const v8h*)(ap + 16 + kb8);
        }
        #pragma unroll
        for (int j = 0; j < 2; ++j)
          #pragma unroll
          for (int rb = 0; rb < 2; ++rb)
            acc2[j][rb] = wmma_f16(Aq[rb].v, Bq[j].v, acc2[j][rb]);
      }
      ++sl;
    }

    #pragma unroll
    for (int j = 0; j < 2; ++j) {
      const int col = (nt + j) * 16 + lr;
      #pragma unroll
      for (int rb = 0; rb < 2; ++rb) {
        #pragma unroll
        for (int v = 0; v < 8; ++v) {
          ot[(size_t)(rb * 16 + hi * 8 + v) * ND + col] = acc2[j][rb][v];
        }
      }
    }
  }
}

extern "C" void kernel_launch(void* const* d_in, const int* in_sizes, int n_in,
                              void* d_out, int out_size, void* d_ws, size_t ws_size,
                              hipStream_t stream) {
  (void)in_sizes; (void)n_in; (void)out_size;
  const float* x  = (const float*)d_in[0];
  const float* M  = (const float*)d_in[1];
  const float* W1 = (const float*)d_in[2];
  const float* b1 = (const float*)d_in[3];
  const float* W2 = (const float*)d_in[4];
  const float* b2 = (const float*)d_in[5];
  float* out = (float*)d_out;

  const size_t nPack = (size_t)NH * NH * ND * NK;   // 2,408,448 halves each
  const size_t nX    = (size_t)NH * BS * ND;        // 88,080,384 elements
  _Float16* pW1 = (_Float16*)d_ws;
  _Float16* pW2 = pW1 + nPack;
  float*    gb2 = (float*)(pW2 + nPack);
  _Float16* xh  = (_Float16*)(gb2 + NH * ND);

  const size_t needF16x = nPack * 4 + (size_t)NH * ND * 4 + nX * 2;  // ~186 MB
  const bool useF16x = ws_size >= needF16x;

  const int tpb = 256;
  pack_w1<<<dim3((unsigned)((nPack + tpb - 1) / tpb)), tpb, 0, stream>>>(W1, pW1);
  pack_w2<<<dim3((unsigned)((nPack + tpb - 1) / tpb)), tpb, 0, stream>>>(W2, pW2);
  gate_bias2<<<dim3(NH), tpb, 0, stream>>>(M, b2, gb2);

  dim3 grid(BS / WROWS, NH);                         // 512 x 7 single-wave WGs
  if (useF16x) {
    cvt_x<<<dim3((unsigned)(nX / 8 / tpb)), tpb, 0, stream>>>(x, xh);
    hazard_fused<true><<<grid, 32, 0, stream>>>(x, M, b1, pW1, pW2, gb2, xh, out);
  } else {
    hazard_fused<false><<<grid, 32, 0, stream>>>(x, M, b1, pW1, pW2, gb2,
                                                 (const _Float16*)0, out);
  }
}
